// GATEncoder_24764781429515
// MI455X (gfx1250) — compile-verified, matrix operand order checked by reading, not run on hardware
//
#include <hip/hip_runtime.h>
#include <hip/hip_bf16.h>

typedef __attribute__((ext_vector_type(16))) _Float16 v16h;
typedef __attribute__((ext_vector_type(8)))  float    v8f;

#define GB   32
#define GN   512
#define GD   64
#define GH   8
#define GDK  8

union V16H { v16h v; _Float16 e[16]; };
union V8F  { v8f  v; float    e[8];  };

// 16-bit WMMA operand K mapping (ISA 7.12.2, 16-bit A 16x32):
// lanes 0-15: element e<8 -> K=e, e>=8 -> K=16+(e-8)
// lanes 16-31: element e<8 -> K=8+e, e>=8 -> K=24+(e-8)
__device__ __forceinline__ int kmap(int e, int lane) {
  int khalf = (lane >> 4) << 3;
  return (e < 8) ? (khalf + e) : (16 + khalf + (e - 8));
}

// LDS byte offset of a generic pointer to __shared__ (flat LDS aperture keeps
// the wave-relative offset in addr[31:0], per ISA 10.2 aperture mapping).
__device__ __forceinline__ unsigned lds_off(const void* p) {
  return (unsigned)(unsigned long long)p;
}

__device__ __forceinline__ void async_b128(unsigned lds, const void* g) {
  asm volatile("global_load_async_to_lds_b128 %0, %1, off"
               :: "v"(lds), "v"((unsigned long long)g) : "memory");
}
__device__ __forceinline__ void async_b64(unsigned lds, const void* g) {
  asm volatile("global_load_async_to_lds_b64 %0, %1, off"
               :: "v"(lds), "v"((unsigned long long)g) : "memory");
}

// ---------------------------------------------------------------------------
// Kernel 1: xp = x @ W  (16384 x 64 x 64 GEMM), WMMA f16 -> f32 accumulate.
// One wave per 16-row tile; 4 N-tiles x 2 K-steps = 8 wmma per block.
// ---------------------------------------------------------------------------
__global__ __launch_bounds__(32)
void gat_transform(const float* __restrict__ x, const float* __restrict__ W,
                   float* __restrict__ xp32, _Float16* __restrict__ xp16) {
  const int lane = threadIdx.x;
  const int row0 = blockIdx.x * 16;
  const int r = lane & 15;

  V16H a[2];
  for (int ks = 0; ks < 2; ++ks) {
    const int kb = ks * 32;
    for (int e = 0; e < 16; ++e) {
      int k = kb + kmap(e, lane);
      a[ks].e[e] = (_Float16)x[(row0 + r) * GD + k];
    }
  }
  for (int nt = 0; nt < 4; ++nt) {
    const int col = nt * 16 + r;
    V8F acc;
    for (int q = 0; q < 8; ++q) acc.e[q] = 0.f;
    for (int ks = 0; ks < 2; ++ks) {
      const int kb = ks * 32;
      V16H bm;
      for (int e = 0; e < 16; ++e) {
        int k = kb + kmap(e, lane);
        bm.e[e] = (_Float16)W[k * (GH * GDK) + col];
      }
      acc.v = __builtin_amdgcn_wmma_f32_16x16x32_f16(
          false, a[ks].v, false, bm.v, (short)0, acc.v, false, false);
    }
    for (int rr = 0; rr < 8; ++rr) {
      int m = rr + ((lane >> 4) << 3);
      int rowg = row0 + m;
      float v = acc.e[rr];
      xp32[rowg * GD + col] = v;
      xp16[rowg * GD + col] = (_Float16)v;
    }
  }
}

// ---------------------------------------------------------------------------
// Kernel 2: per-(b,n,h) source/target logit terms  s = xp.a_src, t = xp.a_dst
// ---------------------------------------------------------------------------
__global__ __launch_bounds__(256)
void gat_logits(const float* __restrict__ xp32, const float* __restrict__ a_src,
                const float* __restrict__ a_dst,
                float* __restrict__ s, float* __restrict__ t) {
  int idx = blockIdx.x * 256 + threadIdx.x;  // (bn, h), h fastest
  int h = idx & (GH - 1);
  int bn = idx >> 3;
  float ss = 0.f, tt = 0.f;
  for (int d = 0; d < GDK; ++d) {
    float v = xp32[bn * GD + h * GDK + d];
    ss += v * a_src[h * GDK + d];
    tt += v * a_dst[h * GDK + d];
  }
  s[idx] = ss;
  t[idx] = tt;
}

// ---------------------------------------------------------------------------
// Kernel 3: online softmax stats over sources i per (b, j, h): m and 1/l.
// Coalesced over j within a block; streaming prefetch on the strided adj walk.
// ---------------------------------------------------------------------------
__global__ __launch_bounds__(256)
void gat_stats(const int* __restrict__ adj, const float* __restrict__ s,
               const float* __restrict__ t,
               float* __restrict__ mOut, float* __restrict__ ilOut) {
  int idx = blockIdx.x * 256 + threadIdx.x;  // (b, h, j), j fastest
  int j = idx & (GN - 1);
  int h = (idx >> 9) & (GH - 1);
  int b = idx >> 12;
  float tj = t[(b * GN + j) * GH + h];
  float m = -__builtin_inff(), l = 0.f;
  const int* adjb = adj + b * GN * GN + j;
  for (int i = 0; i < GN; ++i) {
    __builtin_prefetch(adjb + (i + 32) * GN, 0, 0);  // speculative: safe OOB
    bool msk = (adjb[i * GN] != 0) || (i == j);
    if (msk) {
      float e = s[(b * GN + i) * GH + h] + tj;
      e = e > 0.f ? e : 0.2f * e;  // leaky_relu 0.2
      float mn = fmaxf(m, e);
      l = l * __expf(m - mn) + __expf(e - mn);
      m = mn;
    }
  }
  mOut[(b * GN + j) * GH + h] = m;
  ilOut[(b * GN + j) * GH + h] = 1.f / l;
}

// ---------------------------------------------------------------------------
// Kernel 4: aggregation  out[b,j,h,:] = relu(sum_i alpha[i,j,h]*xp[i,h,:] + b)
// Block = (b, 16 targets j); wave w = head h.
//  - s[b,:,:] (16KB) and xp16[b,:,:] (64KB) async-preloaded into LDS
//  - adj tiles double-buffered with global_load_async_to_lds_b64, pipelined
//    against the exp/WMMA compute, synced with s_wait_asynccnt
// ---------------------------------------------------------------------------
__global__ __launch_bounds__(256)
void gat_agg(const int* __restrict__ adj, const float* __restrict__ s,
             const float* __restrict__ t, const float* __restrict__ mIn,
             const float* __restrict__ il, const _Float16* __restrict__ xp16,
             const float* __restrict__ bias, float* __restrict__ out) {
  __shared__ __align__(16) float    sh_s[GN * GH];       // 16 KB
  __shared__ __align__(16) _Float16 sh_x[GN * GD];       // 64 KB
  __shared__ __align__(16) int      sh_adj[2][32 * 16];  // 2 x 2 KB

  const int b   = blockIdx.x >> 5;        // 32 j-tiles per batch
  const int j0  = (blockIdx.x & 31) * 16;
  const int tid = threadIdx.x;
  const int lane = tid & 31;
  const int h  = tid >> 5;                // wave index == head
  const int jl = lane & 15;
  const int jg = j0 + jl;
  const int n  = lane & 15;               // B/D column (d for n<8)

  // async preload: s panel (16KB) + xp16 panel (64KB), in-order per wave
  {
    const char* gs = (const char*)(s + b * GN * GH);
    unsigned ls = lds_off(sh_s);
    for (int q = 0; q < 4; ++q) {
      unsigned off = (unsigned)tid * 16u + (unsigned)q * 4096u;
      async_b128(ls + off, gs + off);
    }
    const char* gx = (const char*)(xp16 + b * GN * GD);
    unsigned lx = lds_off(sh_x);
    for (int q = 0; q < 16; ++q) {
      unsigned off = (unsigned)tid * 16u + (unsigned)q * 4096u;
      async_b128(lx + off, gx + off);
    }
  }
  // adj chunk issue: 32 x 16 ints (2KB) per chunk, b64 per thread
  const int a_irow = tid >> 3;        // 0..31
  const int a_jc2  = (tid & 7) * 2;   // even columns
  {
    const int* g = adj + (b * GN + a_irow) * GN + j0 + a_jc2;
    async_b64(lds_off(&sh_adj[0][a_irow * 16 + a_jc2]), g);
  }

  const float tj = t  [(b * GN + jg) * GH + h];
  const float mj = mIn[(b * GN + jg) * GH + h];
  const float iv = il [(b * GN + jg) * GH + h];

  V8F acc;
  for (int q = 0; q < 8; ++q) acc.e[q] = 0.f;

  for (int kc = 0; kc < 16; ++kc) {
    const int ibase = kc * 32;
    if (kc < 15) {
      const int* g = adj + (b * GN + ibase + 32 + a_irow) * GN + j0 + a_jc2;
      async_b64(lds_off(&sh_adj[(kc + 1) & 1][a_irow * 16 + a_jc2]), g);
      asm volatile("s_wait_asynccnt 0x1" ::: "memory");
    } else {
      asm volatile("s_wait_asynccnt 0x0" ::: "memory");
    }
    __syncthreads();  // chunk kc (and preloads) visible to all waves

    const int* am = sh_adj[kc & 1];
    V16H A, Bm;
    for (int e = 0; e < 16; ++e) {
      int io = kmap(e, lane);
      int ig = ibase + io;
      float ev = sh_s[ig * GH + h] + tj;
      ev = ev > 0.f ? ev : 0.2f * ev;
      bool msk = (am[io * 16 + jl] != 0) || (ig == jg);
      float p = msk ? (__expf(ev - mj) * iv) : 0.f;
      A.e[e] = (_Float16)p;
      Bm.e[e] = (n < GDK) ? sh_x[ig * GD + h * GDK + n] : (_Float16)0.f;
    }
    acc.v = __builtin_amdgcn_wmma_f32_16x16x32_f16(
        false, A.v, false, Bm.v, (short)0, acc.v, false, false);
    __syncthreads();  // all reads of this adj buffer done before reuse
  }

  if (n < GDK) {
    for (int r = 0; r < 8; ++r) {
      int m = r + ((lane >> 4) << 3);
      float v = acc.e[r] + bias[h * GDK + n];
      out[(b * GN + (j0 + m)) * GD + h * GDK + n] = v > 0.f ? v : 0.f;
    }
  }
}

// ---------------------------------------------------------------------------
extern "C" void kernel_launch(void* const* d_in, const int* in_sizes, int n_in,
                              void* d_out, int out_size, void* d_ws, size_t ws_size,
                              hipStream_t stream) {
  (void)in_sizes; (void)n_in; (void)out_size; (void)ws_size;
  const float* x    = (const float*)d_in[0];
  const int*   adj  = (const int*)  d_in[1];
  const float* W1   = (const float*)d_in[2];
  const float* as1  = (const float*)d_in[3];
  const float* ad1  = (const float*)d_in[4];
  const float* b1   = (const float*)d_in[5];
  const float* W2   = (const float*)d_in[6];
  const float* as2  = (const float*)d_in[7];
  const float* ad2  = (const float*)d_in[8];
  const float* b2   = (const float*)d_in[9];

  char* ws = (char*)d_ws;                       // ~12 MB layout
  float*     xp32 = (float*)    (ws + 0);                 // 4 MB
  _Float16*  xp16 = (_Float16*) (ws + (4u  << 20));       // 2 MB
  float*     s    = (float*)    (ws + (6u  << 20));       // 512 KB
  float*     t    = (float*)    (ws + (6u  << 20) + (512u << 10));
  float*     m    = (float*)    (ws + (7u  << 20));
  float*     il   = (float*)    (ws + (7u  << 20) + (512u << 10));
  float*     out1 = (float*)    (ws + (8u  << 20));       // 4 MB

  const int gTiles  = (GB * GN) / 16;        // 1024
  const int gLogits = (GB * GN * GH) / 256;  // 512
  const int gStats  = (GB * GH * GN) / 256;  // 512
  const int gAgg    = GB * (GN / 16);        // 1024

  // Layer 1
  gat_transform<<<gTiles, 32, 0, stream>>>(x, W1, xp32, xp16);
  gat_logits<<<gLogits, 256, 0, stream>>>(xp32, as1, ad1, s, t);
  gat_stats<<<gStats, 256, 0, stream>>>(adj, s, t, m, il);
  gat_agg<<<gAgg, 256, 0, stream>>>(adj, s, t, m, il, xp16, b1, out1);
  // Layer 2
  gat_transform<<<gTiles, 32, 0, stream>>>(out1, W2, xp32, xp16);
  gat_logits<<<gLogits, 256, 0, stream>>>(xp32, as2, ad2, s, t);
  gat_stats<<<gStats, 256, 0, stream>>>(adj, s, t, m, il);
  gat_agg<<<gAgg, 256, 0, stream>>>(adj, s, t, m, il, xp16, b2, (float*)d_out);
}
